// MeMNN_44933947851210
// MI455X (gfx1250) — compile-verified, matrix operand order checked by reading, not run on hardware
//
#include <hip/hip_runtime.h>
#include <hip/hip_bf16.h>
#include <math.h>

#define NUM_HOP 3
#define VOCAB   32000
#define D       512
#define MEM     60
#define BS      64
#define STORY   50
#define SENT    32
#define QLEN    16

typedef __attribute__((ext_vector_type(16))) __bf16 v16bf;
typedef __attribute__((ext_vector_type(8)))  __bf16 v8bf;
typedef __attribute__((ext_vector_type(8)))  float  v8f;
typedef __attribute__((ext_vector_type(4)))  float  v4f;   // clang ext-vector: OK for nontemporal builtin

// ---------------------------------------------------------------------------
// u = sum_q A0[query[b,q]]  (skip padding idx 0)
// grid: BS blocks, 128 threads, each thread owns 4 consecutive dims (float4)
// ---------------------------------------------------------------------------
__global__ void memnn_query_encode(const int* __restrict__ query,
                                   const float* __restrict__ A0,
                                   float* __restrict__ u) {
    const int b  = blockIdx.x;
    const int d4 = threadIdx.x * 4;
    float4 acc = make_float4(0.f, 0.f, 0.f, 0.f);
    for (int q = 0; q < QLEN; ++q) {
        const int idx = query[b * QLEN + q];
        if (idx != 0) {
            const float4 e = *reinterpret_cast<const float4*>(A0 + (size_t)idx * D + d4);
            acc.x += e.x; acc.y += e.y; acc.z += e.z; acc.w += e.w;
        }
    }
    *reinterpret_cast<float4*>(u + (size_t)b * D + d4) = acc;
}

// ---------------------------------------------------------------------------
// Fused per-hop memory/output embedding:
//   m[b,s,:] = sum_w pe[w,:] * Am[ctx[b,s,w]]  + T_A[s,:]
//   c[b,s,:] = sum_w          Ac[ctx[b,s,w]]
// grid: BS*STORY blocks, 128 threads (float4 per thread)
// ---------------------------------------------------------------------------
__global__ void memnn_embed_mc(const int* __restrict__ ctx,
                               const float* __restrict__ Am,
                               const float* __restrict__ Ac,
                               const float* __restrict__ TA,
                               float* __restrict__ m,
                               float* __restrict__ c) {
    const int bs = blockIdx.x;            // b*STORY + s
    const int s  = bs % STORY;
    const int d4 = threadIdx.x * 4;

    // pe[w, k-1] = 1 - i/J - (k/d)*(1 - 2i/J), i = w+1, k = dim+1
    float kf[4];
#pragma unroll
    for (int j = 0; j < 4; ++j) kf[j] = (float)(d4 + j + 1) / (float)D;

    float4 macc = make_float4(0.f, 0.f, 0.f, 0.f);
    float4 cacc = make_float4(0.f, 0.f, 0.f, 0.f);
    const int* cw = ctx + (size_t)bs * SENT;

    for (int w = 0; w < SENT; ++w) {
        const int idx = cw[w];
        if (idx == 0) continue;           // padding row is zero
        const float iw    = (float)(w + 1) / (float)SENT;
        const float base  = 1.0f - iw;
        const float slope = 1.0f - 2.0f * iw;
        const float4 em = *reinterpret_cast<const float4*>(Am + (size_t)idx * D + d4);
        const float4 ec = *reinterpret_cast<const float4*>(Ac + (size_t)idx * D + d4);
        macc.x = fmaf(em.x, base - kf[0] * slope, macc.x);
        macc.y = fmaf(em.y, base - kf[1] * slope, macc.y);
        macc.z = fmaf(em.z, base - kf[2] * slope, macc.z);
        macc.w = fmaf(em.w, base - kf[3] * slope, macc.w);
        cacc.x += ec.x; cacc.y += ec.y; cacc.z += ec.z; cacc.w += ec.w;
    }

    // temporal encoding (memory_size > story_size branch: first STORY rows)
    const float4 t = *reinterpret_cast<const float4*>(TA + (size_t)s * D + d4);
    macc.x += t.x; macc.y += t.y; macc.z += t.z; macc.w += t.w;

    *reinterpret_cast<float4*>(m + (size_t)bs * D + d4) = macc;
    *reinterpret_cast<float4*>(c + (size_t)bs * D + d4) = cacc;
}

// ---------------------------------------------------------------------------
// Attention + update:  p = softmax(m·u);  u += p·c
// grid: BS blocks, 256 threads (8 wave32's)
// ---------------------------------------------------------------------------
__global__ void memnn_attn(const float* __restrict__ m,
                           const float* __restrict__ c,
                           float* __restrict__ u) {
    __shared__ float su[D];
    __shared__ float sc[STORY];
    const int b   = blockIdx.x;
    const int tid = threadIdx.x;

    for (int d = tid; d < D; d += 256) su[d] = u[(size_t)b * D + d];
    __syncthreads();

    const int wave = tid >> 5;
    const int lane = tid & 31;
    for (int s = wave; s < STORY; s += 8) {
        const float* mr = m + (size_t)(b * STORY + s) * D;
        float acc = 0.f;
        for (int d = lane; d < D; d += 32) acc += mr[d] * su[d];
#pragma unroll
        for (int off = 16; off > 0; off >>= 1) acc += __shfl_xor(acc, off, 32);
        if (lane == 0) sc[s] = acc;
    }
    __syncthreads();

    // softmax over STORY (redundantly per thread; 50 elems is cheap)
    float mx = -INFINITY;
    for (int s = 0; s < STORY; ++s) mx = fmaxf(mx, sc[s]);
    float sum = 0.f;
    for (int s = 0; s < STORY; ++s) sum += expf(sc[s] - mx);
    const float inv = 1.0f / sum;
    __syncthreads();
    if (tid < STORY) sc[tid] = expf(sc[tid] - mx) * inv;
    __syncthreads();

    for (int d = tid; d < D; d += 256) {
        float acc = su[d];
        for (int s = 0; s < STORY; ++s)
            acc = fmaf(sc[s], c[(size_t)(b * STORY + s) * D + d], acc);
        u[(size_t)b * D + d] = acc;
    }
}

// ---------------------------------------------------------------------------
// One-shot f32 -> bf16 conversion of u (64x512, row-major).
// grid: BS blocks, 128 threads, 4 elems/thread.
// ---------------------------------------------------------------------------
__global__ void memnn_u_to_bf16(const float* __restrict__ u,
                                __bf16* __restrict__ ubf) {
    const int i = (blockIdx.x * 128 + threadIdx.x) * 4;
    const float4 f = *reinterpret_cast<const float4*>(u + i);
    __bf16 o[4] = { (__bf16)f.x, (__bf16)f.y, (__bf16)f.z, (__bf16)f.w };
    *reinterpret_cast<ulonglong1*>(ubf + i) = *reinterpret_cast<ulonglong1*>(o);
}

// ---------------------------------------------------------------------------
// logits = u @ A3^T via v_wmma_f32_16x16x32_bf16 (bf16 in, f32 accumulate).
//
// Mapping: 1 wave owns one 16-wide N-tile and computes ALL four 16-row
// M-tiles (M=64) with 4 accumulators, so the B-fragment (A3 rows, converted
// f32->bf16 once per K-step) is reused in registers across 4 WMMAs and A[3]
// is streamed from memory exactly once (NT loads: keep the L2 for the
// embedding-gather working set).  A-fragments come from the pre-converted
// 64KB u_bf16 buffer: two 16B v8bf loads + shufflevector, no per-tile cvt.
//
// Fragment layouts (ISA 7.12.2, wave32):
//   A: lane r (0..15) holds row M=r; lane halves hold K {0..7,16..23} vs
//      {8..15,24..31}.  B column n = A3 row n, same per-lane K pattern.
//   D: VGPR v -> lanes 0-15: (row v, col lane), lanes 16-31: (row v+8, ...).
// Vocab row 0 is the zeroed padding row -> branchless mask (EXEC stays
// all-ones across the WMMAs, as required).
//
// grid: VOCAB/16/8 = 250 blocks, 256 threads (8 waves = 8 N-tiles/block).
// ---------------------------------------------------------------------------
__global__ void memnn_logits_wmma(const __bf16* __restrict__ ubf,
                                  const float* __restrict__ A3,
                                  float* __restrict__ out) {
    const int wave = threadIdx.x >> 5;
    const int lane = threadIdx.x & 31;
    const int n0   = (blockIdx.x * 8 + wave) * 16;
    const int half = lane >> 4;
    const int r    = lane & 15;
    const int koff = half * 8;

    const int    vrow = n0 + r;
    const float* arow = A3 + (size_t)vrow * D;
    const float  bm   = (vrow == 0) ? 0.0f : 1.0f;   // padding row mask

    v8f acc0 = {0.f,0.f,0.f,0.f,0.f,0.f,0.f,0.f};
    v8f acc1 = acc0, acc2 = acc0, acc3 = acc0;

    for (int kt = 0; kt < D / 32; ++kt) {
        const int kb = kt * 32 + koff;

        // B fragment: stream A3 row once (non-temporal), convert to bf16
        const v4f b0 = __builtin_nontemporal_load(reinterpret_cast<const v4f*>(arow + kb));
        const v4f b1 = __builtin_nontemporal_load(reinterpret_cast<const v4f*>(arow + kb + 4));
        const v4f b2 = __builtin_nontemporal_load(reinterpret_cast<const v4f*>(arow + kb + 16));
        const v4f b3 = __builtin_nontemporal_load(reinterpret_cast<const v4f*>(arow + kb + 20));
        v16bf bf;
        bf[0]  = (__bf16)(b0[0] * bm); bf[1]  = (__bf16)(b0[1] * bm);
        bf[2]  = (__bf16)(b0[2] * bm); bf[3]  = (__bf16)(b0[3] * bm);
        bf[4]  = (__bf16)(b1[0] * bm); bf[5]  = (__bf16)(b1[1] * bm);
        bf[6]  = (__bf16)(b1[2] * bm); bf[7]  = (__bf16)(b1[3] * bm);
        bf[8]  = (__bf16)(b2[0] * bm); bf[9]  = (__bf16)(b2[1] * bm);
        bf[10] = (__bf16)(b2[2] * bm); bf[11] = (__bf16)(b2[3] * bm);
        bf[12] = (__bf16)(b3[0] * bm); bf[13] = (__bf16)(b3[1] * bm);
        bf[14] = (__bf16)(b3[2] * bm); bf[15] = (__bf16)(b3[3] * bm);

        // A fragments for the four M-tiles from pre-converted bf16 u
        const __bf16* up0 = ubf + (size_t)(0 * 16 + r) * D + kb;
        const __bf16* up1 = ubf + (size_t)(1 * 16 + r) * D + kb;
        const __bf16* up2 = ubf + (size_t)(2 * 16 + r) * D + kb;
        const __bf16* up3 = ubf + (size_t)(3 * 16 + r) * D + kb;

        v8bf a0l = *reinterpret_cast<const v8bf*>(up0);
        v8bf a0h = *reinterpret_cast<const v8bf*>(up0 + 16);
        v16bf af0 = __builtin_shufflevector(a0l, a0h, 0,1,2,3,4,5,6,7,8,9,10,11,12,13,14,15);
        acc0 = __builtin_amdgcn_wmma_f32_16x16x32_bf16(false, af0, false, bf, (short)0, acc0, false, false);

        v8bf a1l = *reinterpret_cast<const v8bf*>(up1);
        v8bf a1h = *reinterpret_cast<const v8bf*>(up1 + 16);
        v16bf af1 = __builtin_shufflevector(a1l, a1h, 0,1,2,3,4,5,6,7,8,9,10,11,12,13,14,15);
        acc1 = __builtin_amdgcn_wmma_f32_16x16x32_bf16(false, af1, false, bf, (short)0, acc1, false, false);

        v8bf a2l = *reinterpret_cast<const v8bf*>(up2);
        v8bf a2h = *reinterpret_cast<const v8bf*>(up2 + 16);
        v16bf af2 = __builtin_shufflevector(a2l, a2h, 0,1,2,3,4,5,6,7,8,9,10,11,12,13,14,15);
        acc2 = __builtin_amdgcn_wmma_f32_16x16x32_bf16(false, af2, false, bf, (short)0, acc2, false, false);

        v8bf a3l = *reinterpret_cast<const v8bf*>(up3);
        v8bf a3h = *reinterpret_cast<const v8bf*>(up3 + 16);
        v16bf af3 = __builtin_shufflevector(a3l, a3h, 0,1,2,3,4,5,6,7,8,9,10,11,12,13,14,15);
        acc3 = __builtin_amdgcn_wmma_f32_16x16x32_bf16(false, af3, false, bf, (short)0, acc3, false, false);
    }

    const int col = n0 + r;
#pragma unroll
    for (int v = 0; v < 8; ++v) {
        const int row = v + half * 8;
        out[(size_t)(row +  0) * VOCAB + col] = acc0[v];
        out[(size_t)(row + 16) * VOCAB + col] = acc1[v];
        out[(size_t)(row + 32) * VOCAB + col] = acc2[v];
        out[(size_t)(row + 48) * VOCAB + col] = acc3[v];
    }
}

// ---------------------------------------------------------------------------
// Row softmax over VOCAB (in place on d_out). grid: BS blocks, 256 threads.
// ---------------------------------------------------------------------------
__global__ void memnn_softmax_rows(float* __restrict__ out) {
    __shared__ float red[256];
    const int b   = blockIdx.x;
    const int tid = threadIdx.x;
    float* row = out + (size_t)b * VOCAB;

    float mx = -INFINITY;
    for (int v = tid; v < VOCAB; v += 256) mx = fmaxf(mx, row[v]);
    red[tid] = mx; __syncthreads();
    for (int s = 128; s > 0; s >>= 1) {
        if (tid < s) red[tid] = fmaxf(red[tid], red[tid + s]);
        __syncthreads();
    }
    mx = red[0]; __syncthreads();

    float sum = 0.f;
    for (int v = tid; v < VOCAB; v += 256) sum += expf(row[v] - mx);
    red[tid] = sum; __syncthreads();
    for (int s = 128; s > 0; s >>= 1) {
        if (tid < s) red[tid] += red[tid + s];
        __syncthreads();
    }
    const float inv = 1.0f / red[0];

    for (int v = tid; v < VOCAB; v += 256) row[v] = expf(row[v] - mx) * inv;
}

// ---------------------------------------------------------------------------
extern "C" void kernel_launch(void* const* d_in, const int* in_sizes, int n_in,
                              void* d_out, int out_size, void* d_ws, size_t ws_size,
                              hipStream_t stream) {
    (void)in_sizes; (void)n_in; (void)out_size; (void)ws_size;

    const int*   ctx = (const int*)d_in[0];     // (BS, STORY, SENT)
    const int*   qry = (const int*)d_in[1];     // (BS, QLEN)
    const float* A   = (const float*)d_in[2];   // (NUM_HOP+1, VOCAB, D)
    const float* TA  = (const float*)d_in[3];   // (1, MEM, D)
    float*       out = (float*)d_out;           // (BS, VOCAB)

    // workspace layout (floats): u | m | c | u_bf16
    float*  u   = (float*)d_ws;
    float*  m   = u + (size_t)BS * D;
    float*  c   = m + (size_t)BS * STORY * D;
    __bf16* ubf = (__bf16*)(c + (size_t)BS * STORY * D);

    const size_t table = (size_t)VOCAB * D;

    memnn_query_encode<<<BS, 128, 0, stream>>>(qry, A /* A[0] */, u);

    for (int h = 0; h < NUM_HOP; ++h) {
        memnn_embed_mc<<<BS * STORY, 128, 0, stream>>>(
            ctx, A + (size_t)h * table, A + (size_t)(h + 1) * table, TA, m, c);
        memnn_attn<<<BS, 256, 0, stream>>>(m, c, u);
    }

    memnn_u_to_bf16<<<BS, 128, 0, stream>>>(u, ubf);
    memnn_logits_wmma<<<VOCAB / 16 / 8, 256, 0, stream>>>(
        ubf, A + (size_t)NUM_HOP * table, out);
    memnn_softmax_rows<<<BS, 256, 0, stream>>>(out);
}
